// MultiHeadedAttention_37005438222867
// MI455X (gfx1250) — matrix operand order repaired and checked by reading, not compiled, not run on hardware
//
#include <hip/hip_runtime.h>

#define B_  4
#define S_  2048
#define DM_ 1024
#define H_  16
#define DK_ 64

typedef __bf16 bf16;
typedef bf16  v16bf __attribute__((ext_vector_type(16)));
typedef bf16  v8bf  __attribute__((ext_vector_type(8)));
typedef float v8f   __attribute__((ext_vector_type(8)));
typedef unsigned int v4ui __attribute__((ext_vector_type(4)));
typedef int v8i_ __attribute__((ext_vector_type(8)));
typedef int v4i_ __attribute__((ext_vector_type(4)));

union FragBF { v16bf v; v8bf h[2]; };

// raw hardware base-2 exponential (v_exp_f32)
__device__ __forceinline__ float fexp2(float x) { return __builtin_amdgcn_exp2f(x); }

// fp32 -> bf16, round-to-nearest-even, via bit ops
__device__ __forceinline__ bf16 f2bf(float x) {
  unsigned u = __float_as_uint(x);
  u += 0x7FFFu + ((u >> 16) & 1u);
  unsigned short s = (unsigned short)(u >> 16);
  return __builtin_bit_cast(bf16, s);
}

// ---- WMMA fragment loaders (CDNA5 ISA 7.12.2 layouts, wave32) ----
// A (16x32, MxK): lane l<16: row M=l, K={k0..k0+7, k0+16..k0+23}; l>=16: K shifted by 8.
__device__ __forceinline__ v16bf load_a_frag(const bf16* M, int row0, int k0, int ld) {
  int lane = threadIdx.x & 31;
  const bf16* p = M + (size_t)(row0 + (lane & 15)) * ld + k0 + ((lane >> 4) << 3);
  FragBF f;
  f.h[0] = *(const v8bf*)(p);
  f.h[1] = *(const v8bf*)(p + 16);
  return f.v;
}
// B (32x16, KxN) for NT product: lane l holds column n=l&15 (= row n of a row-major
// matrix); K = {k0..k0+7, k0+16..k0+23}, l>=16 shifted by 8. Works for global or LDS ptrs.
__device__ __forceinline__ v16bf load_b_frag(const bf16* M, int row0, int k0, int ld) {
  int lane = threadIdx.x & 31;
  const bf16* p = M + (size_t)(row0 + (lane & 15)) * ld + k0 + ((lane >> 4) << 3);
  FragBF f;
  f.h[0] = *(const v8bf*)(p);
  f.h[1] = *(const v8bf*)(p + 16);
  return f.v;
}

#define WMMA_BF16(a, b, c) \
  __builtin_amdgcn_wmma_f32_16x16x32_bf16(false, (a), false, (b), (short)0, (c), false, false)

// ---- Tensor Data Mover: 2D tile load Global->LDS (ISA ch.8 D# layout) ----
__device__ __forceinline__ unsigned lds_addr_of(const void* p) {
  return (unsigned)(unsigned long long)(size_t)p;  // addr[31:0] = LDS byte offset
}

__device__ __forceinline__ void tdm_load_2d(void* lds_ptr, const void* gptr,
                                            unsigned tw, unsigned th,
                                            unsigned stride, unsigned dsz_code) {
  unsigned long long ga = (unsigned long long)(size_t)gptr;
  v4ui g0;
  g0[0] = 1u;                                                  // count=1
  g0[1] = lds_addr_of(lds_ptr);                                // lds_addr
  g0[2] = (unsigned)ga;                                        // global_addr[31:0]
  g0[3] = ((unsigned)(ga >> 32) & 0x01FFFFFFu) | 0x80000000u;  // addr[56:32] | type=2
  v8i_ g1;
  g1[0] = (int)(dsz_code << 16);   // data_size
  g1[1] = (int)(tw << 16);         // tensor_dim0[15:0]
  g1[2] = (int)(th << 16);         // tensor_dim1[15:0]
  g1[3] = (int)(tw << 16);         // tile_dim0
  g1[4] = (int)th;                 // tile_dim1 (tile_dim2=0)
  g1[5] = (int)stride;             // tensor_dim0_stride[31:0]
  g1[6] = 0;
  g1[7] = 0;
  v4i_ z4 = {0, 0, 0, 0};
#if __clang_major__ >= 23
  v8i_ z8 = {0, 0, 0, 0, 0, 0, 0, 0};
  __builtin_amdgcn_tensor_load_to_lds(g0, g1, z4, z4, z8, 0);
#else
  __builtin_amdgcn_tensor_load_to_lds(g0, g1, z4, z4, 0);
#endif
}

// ---- elementwise fp32 -> bf16 conversion ----
__global__ void cvt_f32_to_bf16(const float* __restrict__ src, bf16* __restrict__ dst, int n) {
  int i = blockIdx.x * blockDim.x + threadIdx.x;
  if (i < n) dst[i] = f2bf(src[i]);
}

__global__ void zero_f32(float* __restrict__ p, int n) {
  int i = blockIdx.x * blockDim.x + threadIdx.x;
  if (i < n) p[i] = 0.0f;
}

// ---- pack int32 mask -> 1 bit per element (wave32 ballot) ----
__global__ void pack_mask(const int* __restrict__ mask, unsigned* __restrict__ bits, int n) {
  int t = blockIdx.x * blockDim.x + threadIdx.x;
  int pred = (t < n) ? (mask[t] != 0) : 0;
  unsigned w = (unsigned)__ballot(pred);
  if ((threadIdx.x & 31) == 0 && t < n) bits[t >> 5] = w;
}

// ---- C[m,n] = sum_d A[m,d]*B[n,d] + bias[n] (NT GEMM, bf16 in/out) ----
// 4 waves; block tile 128M x 64N; wave tile 32M x 64N. Shared 64x32 B tile
// TDM-staged into double-buffered LDS by wave 0; A frags software-pipelined.
__global__ __launch_bounds__(128) void gemm_nt_bias_bf16(
    const bf16* __restrict__ A, const bf16* __restrict__ Bm,
    const float* __restrict__ bias, bf16* __restrict__ C, int Kdim, int Ndim) {
  __shared__ bf16 bbuf[2][64 * 32];

  int lane = threadIdx.x & 31;
  int wave = threadIdx.x >> 5;
  int m0 = blockIdx.x * 128 + wave * 32;
  int n0 = blockIdx.y * 64;

  if (wave == 0)
    tdm_load_2d(bbuf[0], Bm + (size_t)n0 * Kdim, 32, 64, Kdim, 1);

  v16bf a0 = load_a_frag(A, m0, 0, Kdim);
  v16bf a1 = load_a_frag(A, m0 + 16, 0, Kdim);
  v8f acc[8] = {};

  const int nsteps = Kdim >> 5;
  for (int t = 0; t < nsteps; ++t) {
    int k = t << 5;
    __builtin_amdgcn_s_wait_tensorcnt(0);
    __syncthreads();
    if (wave == 0 && t + 1 < nsteps)
      tdm_load_2d(bbuf[(t + 1) & 1], Bm + (size_t)n0 * Kdim + (k + 32), 32, 64, Kdim, 1);

    int kn = (t + 1 < nsteps) ? (k + 32) : 0;
    v16bf na0 = load_a_frag(A, m0, kn, Kdim);
    v16bf na1 = load_a_frag(A, m0 + 16, kn, Kdim);

    const bf16* cur = bbuf[t & 1];
#pragma unroll
    for (int j = 0; j < 4; ++j) {
      v16bf b = load_b_frag(cur, j * 16, 0, 32);
      acc[j]     = WMMA_BF16(a0, b, acc[j]);
      acc[4 + j] = WMMA_BF16(a1, b, acc[4 + j]);
    }
    a0 = na0; a1 = na1;
  }

  int col = lane & 15;
#pragma unroll
  for (int half = 0; half < 2; ++half) {
    int rb = m0 + half * 16 + ((lane >> 4) << 3);
#pragma unroll
    for (int j = 0; j < 4; ++j) {
      int n = n0 + j * 16 + col;
      float bv = bias[n];
#pragma unroll
      for (int i = 0; i < 8; ++i) {
        C[(size_t)(rb + i) * Ndim + n] = f2bf(acc[half * 4 + j][i] + bv);
      }
    }
  }
}

// ---- streaming-softmax attention, scalar value ----
// Block = 8 waves, ONE head, 8 consecutive 16-row q tiles (128 q rows/block):
// the shared 32x64 K tile is TDM-staged once per k-step and consumed by all
// 8 waves (8x less K L2 traffic than per-wave tiles). Mask read as packed bits.
__global__ __launch_bounds__(256) void attn_stream(
    const bf16* __restrict__ Q, const bf16* __restrict__ K,
    const float* __restrict__ value, const unsigned* __restrict__ mbits,
    float* __restrict__ out) {
  __shared__ bf16 kbuf[2][32 * 64];  // 32 K rows x 64 head dims, double buffered

  int lane = threadIdx.x & 31;
  int wave = threadIdx.x >> 5;
  int qchunk = blockIdx.x & 15;        // S/128 = 16 chunks
  int h = (blockIdx.x >> 4) & 15;
  int b = blockIdx.x >> 8;
  int q0 = qchunk * 128 + wave * 16;
  int col = lane & 15;
  int rbase = q0 + ((lane >> 4) << 3);

  const bf16* Qb = Q + (size_t)b * S_ * DM_ + h * DK_;
  const bf16* Kb = K + (size_t)b * S_ * DM_ + h * DK_;
  const unsigned* mb = mbits + (size_t)b * S_ * (S_ / 32);
  const float* vb = value + (size_t)b * S_;

  if (wave == 0) tdm_load_2d(kbuf[0], Kb, 64, 32, DM_, 1);

  v16bf a0 = load_a_frag(Qb, q0, 0, DM_);
  v16bf a1 = load_a_frag(Qb, q0, 32, DM_);

  float lmax[8], lden[8], lnum[8];
#pragma unroll
  for (int i = 0; i < 8; ++i) { lmax[i] = -1e30f; lden[i] = 0.0f; lnum[i] = 0.0f; }

  // base-2 domain: exp(x/sqrt(dk)) = exp2(x * log2(e)/8)
  const float scale2 = 0.125f * 1.44269504088896f;
  const int nsteps = S_ / 32;  // 64
  for (int t = 0; t < nsteps; ++t) {
    int k0 = t << 5;
    __builtin_amdgcn_s_wait_tensorcnt(0);
    __syncthreads();
    if (wave == 0 && t + 1 < nsteps)
      tdm_load_2d(kbuf[(t + 1) & 1],
                  (const char*)Kb + (size_t)(k0 + 32) * DM_ * 2, 64, 32, DM_, 1);

    const bf16* kt = kbuf[t & 1];
    // scores for K rows [k0, k0+16) and [k0+16, k0+32)
    v16bf b00 = load_b_frag(kt, 0, 0, 64);
    v16bf b01 = load_b_frag(kt, 0, 32, 64);
    v16bf b10 = load_b_frag(kt, 16, 0, 64);
    v16bf b11 = load_b_frag(kt, 16, 32, 64);
    v8f s0 = {}, s1 = {};
    s0 = WMMA_BF16(a0, b00, s0);
    s0 = WMMA_BF16(a1, b01, s0);
    s1 = WMMA_BF16(a0, b10, s1);
    s1 = WMMA_BF16(a1, b11, s1);

    float vk0 = vb[k0 + col];
    float vk1 = vb[k0 + 16 + col];
#pragma unroll
    for (int i = 0; i < 8; ++i) {
      int row = rbase + i;
      unsigned w = mb[(size_t)row * (S_ / 32) + (k0 >> 5)];  // 32 mask bits
      int mv0 = (w >> col) & 1;
      int mv1 = (w >> (16 + col)) & 1;

      float sc = s0[i] * scale2;
      float m = fmaxf(lmax[i], sc);
      float c = fexp2(lmax[i] - m);
      float e = mv0 ? fexp2(sc - m) : 0.0f;
      lden[i] = lden[i] * c + e;
      lnum[i] = lnum[i] * c + e * vk0;
      lmax[i] = m;

      sc = s1[i] * scale2;
      m = fmaxf(lmax[i], sc);
      c = fexp2(lmax[i] - m);
      e = mv1 ? fexp2(sc - m) : 0.0f;
      lden[i] = lden[i] * c + e;
      lnum[i] = lnum[i] * c + e * vk1;
      lmax[i] = m;
    }
  }

  // merge 16 lane-local partials per half-wave (butterfly over lanes 0..15)
#pragma unroll
  for (int i = 0; i < 8; ++i) {
    float m = lmax[i], d = lden[i], n = lnum[i];
#pragma unroll
    for (int off = 1; off < 16; off <<= 1) {
      float m2 = __shfl_xor(m, off);
      float d2 = __shfl_xor(d, off);
      float n2 = __shfl_xor(n, off);
      float M = fmaxf(m, m2);
      float c1 = fexp2(m - M), c2 = fexp2(m2 - M);
      d = d * c1 + d2 * c2;
      n = n * c1 + n2 * c2;
      m = M;
    }
    lden[i] = d; lnum[i] = n;
  }

  if (col == 0) {
#pragma unroll
    for (int i = 0; i < 8; ++i) {
      float r = (lnum[i] / lden[i]) * (1.0f / (float)H_);
      atomicAdd(&out[(size_t)b * S_ + rbase + i], r);
    }
  }
}

extern "C" void kernel_launch(void* const* d_in, const int* in_sizes, int n_in,
                              void* d_out, int out_size, void* d_ws, size_t ws_size,
                              hipStream_t stream) {
  (void)in_sizes; (void)n_in; (void)ws_size;
  const float* query = (const float*)d_in[0];
  const float* key   = (const float*)d_in[1];
  const float* value = (const float*)d_in[2];
  const int*   mask  = (const int*)d_in[3];
  const float* Wq    = (const float*)d_in[4];
  const float* bq    = (const float*)d_in[5];
  const float* Wk    = (const float*)d_in[6];
  const float* bk    = (const float*)d_in[7];
  float* out = (float*)d_out;

  char* ws = (char*)d_ws;
  const size_t nX = (size_t)B_ * S_ * DM_;  // 8388608
  const size_t nW = (size_t)DM_ * DM_;      // 1048576
  bf16* Xq  = (bf16*)(ws);
  bf16* Xk  = (bf16*)(ws + 2 * nX);
  bf16* Wqb = (bf16*)(ws + 4 * nX);
  bf16* Wkb = (bf16*)(ws + 4 * nX + 2 * nW);
  bf16* Qb  = (bf16*)(ws + 4 * nX + 4 * nW);
  bf16* Kb  = (bf16*)(ws + 6 * nX + 4 * nW);
  unsigned* Mb = (unsigned*)(ws + 8 * nX + 4 * nW);  // 2 MB packed mask bits

  cvt_f32_to_bf16<<<(int)((nX + 255) / 256), 256, 0, stream>>>(query, Xq, (int)nX);
  cvt_f32_to_bf16<<<(int)((nX + 255) / 256), 256, 0, stream>>>(key,   Xk, (int)nX);
  cvt_f32_to_bf16<<<(int)((nW + 255) / 256), 256, 0, stream>>>(Wq,    Wqb, (int)nW);
  cvt_f32_to_bf16<<<(int)((nW + 255) / 256), 256, 0, stream>>>(Wk,    Wkb, (int)nW);

  const int nM = B_ * S_ * S_;  // 16777216
  pack_mask<<<nM / 256, 256, 0, stream>>>(mask, Mb, nM);

  dim3 ggrid(B_ * S_ / 128, DM_ / 64);
  gemm_nt_bias_bf16<<<ggrid, 128, 0, stream>>>(Xq, Wqb, bq, Qb, DM_, DM_);
  gemm_nt_bias_bf16<<<ggrid, 128, 0, stream>>>(Xk, Wkb, bk, Kb, DM_, DM_);

  zero_f32<<<(out_size + 255) / 256, 256, 0, stream>>>(out, out_size);
  // blocks: b(4) x h(16) x qchunk(16) = 1024, 8 waves each
  attn_stream<<<B_ * H_ * (S_ / 128), 256, 0, stream>>>(Qb, Kb, value, Mb, out);
}
